// EDTAttention_60739427500184
// MI455X (gfx1250) — matrix operand order repaired and checked by reading, not compiled, not run on hardware
//
#include <hip/hip_runtime.h>
#include <math.h>

// Problem shape (fixed by reference setup_inputs)
#define B_DIM  16384
#define DK_DIM 512
#define NK_DIM 4096
#define DV_DIM 512
#define H_DIM  32

typedef __attribute__((ext_vector_type(16))) __bf16 v16bf;
typedef __attribute__((ext_vector_type(8)))  __bf16 v8bf;
typedef __attribute__((ext_vector_type(8)))  float  v8f;
typedef int v4i __attribute__((ext_vector_type(4)));

// ---------------- CDNA5 async global->LDS path (guarded) ----------------

#if defined(__has_builtin)
#if __has_builtin(__builtin_amdgcn_global_load_async_to_lds_b128) && \
    __has_builtin(__builtin_amdgcn_s_wait_asynccnt)
#define EDT_ASYNC 1
#endif
#endif
#ifndef EDT_ASYNC
#define EDT_ASYNC 0
#endif

__device__ __forceinline__ void cp_async_f4(float* lds_dst, const float* gsrc) {
#if EDT_ASYNC
  __builtin_amdgcn_global_load_async_to_lds_b128(
      (v4i __attribute__((address_space(1)))*)gsrc,
      (v4i __attribute__((address_space(3)))*)lds_dst,
      0, 0);
#else
  *(float4*)lds_dst = *(const float4*)gsrc;
#endif
}

template <int N>
__device__ __forceinline__ void wait_async_le() {
#if EDT_ASYNC
  __builtin_amdgcn_s_wait_asynccnt(N);
#endif
}

// ---------------- fragment helpers ----------------

// Build a v16bf A/B fragment from two 8-float contiguous LDS chunks (fp32 tile).
__device__ __forceinline__ v16bf frag_from_f32(const float* p0, const float* p1) {
  const float4 a0 = *(const float4*)p0;
  const float4 a1 = *(const float4*)(p0 + 4);
  const float4 b0 = *(const float4*)p1;
  const float4 b1 = *(const float4*)(p1 + 4);
  v16bf r;
  r[0]  = (__bf16)a0.x; r[1]  = (__bf16)a0.y; r[2]  = (__bf16)a0.z; r[3]  = (__bf16)a0.w;
  r[4]  = (__bf16)a1.x; r[5]  = (__bf16)a1.y; r[6]  = (__bf16)a1.z; r[7]  = (__bf16)a1.w;
  r[8]  = (__bf16)b0.x; r[9]  = (__bf16)b0.y; r[10] = (__bf16)b0.z; r[11] = (__bf16)b0.w;
  r[12] = (__bf16)b1.x; r[13] = (__bf16)b1.y; r[14] = (__bf16)b1.z; r[15] = (__bf16)b1.w;
  return r;
}

// Build a v16bf fragment from two contiguous 8-element bf16 LDS chunks.
__device__ __forceinline__ v16bf frag_from_bf16(const __bf16* p0, const __bf16* p1) {
  v8bf a = *(const v8bf*)p0;
  v8bf b = *(const v8bf*)p1;
  return __builtin_shufflevector(a, b, 0,1,2,3,4,5,6,7,8,9,10,11,12,13,14,15);
}

__device__ __forceinline__ float blk_reduce_max(float v, float* red) {
  const int tid = threadIdx.x;
  red[tid] = v; __syncthreads();
  for (int s = 128; s > 0; s >>= 1) {
    if (tid < s) red[tid] = fmaxf(red[tid], red[tid + s]);
    __syncthreads();
  }
  float r = red[0]; __syncthreads();
  return r;
}

__device__ __forceinline__ float blk_reduce_sum(float v, float* red) {
  const int tid = threadIdx.x;
  red[tid] = v; __syncthreads();
  for (int s = 128; s > 0; s >>= 1) {
    if (tid < s) red[tid] += red[tid + s];
    __syncthreads();
  }
  float r = red[0]; __syncthreads();
  return r;
}

// ---------- kernel 1: raw_logits = Q * K^T (async double-buffered, bf16 WMMA) ----------
// Block tile 128(M) x 128(N), K-chunk 32, double-buffered fp32 LDS tiles.
// 8 waves: wave tile 32(M) x 64(N), 8 fp32 accumulators.

#define QK_STEPS (DK_DIM / 32)   // 16

__global__ __launch_bounds__(256) void edt_gemm_qk(
    const float* __restrict__ Q, const float* __restrict__ Ks,
    float* __restrict__ L) {
  __shared__ __align__(16) float lq[2][128][32];   // 32 KB
  __shared__ __align__(16) float lk[2][128][32];   // 32 KB

  const int tid  = threadIdx.x;
  const int lane = tid & 31;
  const int wave = tid >> 5;
  const int half = lane >> 4;
  const int l16  = lane & 15;

  const size_t bm0 = (size_t)blockIdx.y * 128;
  const size_t bn0 = (size_t)blockIdx.x * 128;
  const int wm0 = (wave >> 1) * 32;   // 4 waves along M
  const int wn0 = (wave & 1) * 64;    // 2 waves along N

  v8f acc[2][4] = {};

  auto stage = [&](int b, int k0) {
    // 8 async b128 issues per thread (4 for Q, 4 for K) => ASYNCcnt += 8 / wave
    for (int it = 0; it < 4; ++it) {
      const int idx = tid + it * 256;
      const int row = idx >> 3;
      const int col = (idx & 7) * 4;
      cp_async_f4(&lq[b][row][col], Q + (bm0 + row) * DK_DIM + k0 + col);
      cp_async_f4(&lk[b][row][col], Ks + (bn0 + row) * DK_DIM + k0 + col);
    }
  };

  stage(0, 0);
  for (int kt = 0; kt < QK_STEPS; ++kt) {
    const int b = kt & 1;
    if (kt + 1 < QK_STEPS) {
      stage(b ^ 1, (kt + 1) * 32);   // prefetch next tile into other buffer
      wait_async_le<8>();            // current tile's copies complete (in-order)
    } else {
      wait_async_le<0>();
    }
    __syncthreads();

    v16bf af[2], bfr[4];
    for (int m = 0; m < 2; ++m) {
      const int r = wm0 + m * 16 + l16;
      // A 16x32 bf16 layout: elems 0..7 = K half*8.., elems 8..15 = K 16+half*8..
      af[m] = frag_from_f32(&lq[b][r][half * 8], &lq[b][r][16 + half * 8]);
    }
    for (int n = 0; n < 4; ++n) {
      const int c = wn0 + n * 16 + l16;
      // B 32x16 bf16 layout: elems 0..7 = K half*8.., elems 8..15 = K 16+half*8..
      bfr[n] = frag_from_f32(&lk[b][c][half * 8], &lk[b][c][16 + half * 8]);
    }
    for (int m = 0; m < 2; ++m)
      for (int n = 0; n < 4; ++n)
        acc[m][n] = __builtin_amdgcn_wmma_f32_16x16x32_bf16(
            false, af[m], false, bfr[n], (short)0, acc[m][n], false, false);
    __syncthreads();
  }

  // Store raw logits (fp32). C layout: VGPR r -> M = half*8 + r, N = l16.
  for (int m = 0; m < 2; ++m)
    for (int n = 0; n < 4; ++n) {
      const size_t r0 = bm0 + wm0 + m * 16 + half * 8;
      const size_t c  = bn0 + wn0 + n * 16 + l16;
      for (int r = 0; r < 8; ++r)
        L[(r0 + r) * NK_DIM + c] = acc[m][n][r];
    }
}

// ---------- kernel 2: per-row entropy -> MLP tau -> re-softmax (in place) ----------

__global__ __launch_bounds__(256) void edt_row(
    float* __restrict__ attn, float* __restrict__ entOut, float* __restrict__ tauOut,
    const float* __restrict__ w1, const float* __restrict__ b1,
    const float* __restrict__ w2, const float* __restrict__ b2) {
  __shared__ float buf[NK_DIM];
  __shared__ float red[256];
  __shared__ float s_tau;

  const int tid = threadIdx.x;
  float* rp = attn + (size_t)blockIdx.x * NK_DIM;

  const float base_scale  = 0.04419417382415922f; // 512^-0.5
  const float max_entropy = 8.31776616671934f;    // ln(4096)
  const float EPS = 1e-8f;

  for (int i = tid; i < NK_DIM / 4; i += 256) {
    const float4 v = ((const float4*)rp)[i];
    buf[4 * i + 0] = v.x * base_scale;
    buf[4 * i + 1] = v.y * base_scale;
    buf[4 * i + 2] = v.z * base_scale;
    buf[4 * i + 3] = v.w * base_scale;
  }
  __syncthreads();

  float m = -3.4e38f;
  for (int i = tid; i < NK_DIM; i += 256) m = fmaxf(m, buf[i]);
  m = blk_reduce_max(m, red);

  float z = 0.0f;
  for (int i = tid; i < NK_DIM; i += 256) z += expf(buf[i] - m);
  z = blk_reduce_sum(z, red);
  const float invz = 1.0f / z;

  float es = 0.0f;
  for (int i = tid; i < NK_DIM; i += 256) {
    const float p = expf(buf[i] - m) * invz;
    es += p * logf(p + EPS);
  }
  es = blk_reduce_sum(es, red);
  const float entn = fminf(fmaxf((-es) / (max_entropy + EPS), 0.0f), 1.0f);

  if (tid == 0) {
    float a = 0.0f;
    for (int j = 0; j < H_DIM; ++j) {
      const float h = entn * w1[j] + b1[j];
      const float g = 0.5f * h * (1.0f + erff(h * 0.7071067811865476f));
      a += g * w2[j];
    }
    const float sc = 1.0f / (1.0f + expf(-(a + b2[0])));
    const float tv = 0.1f + 4.9f * sc;
    s_tau = tv;
    entOut[blockIdx.x] = entn;
    tauOut[blockIdx.x] = tv;
  }
  __syncthreads();

  const float itau = 1.0f / (s_tau + EPS);

  float z2 = 0.0f;
  for (int i = tid; i < NK_DIM; i += 256) z2 += expf((buf[i] - m) * itau);
  z2 = blk_reduce_sum(z2, red);
  const float iz2 = 1.0f / z2;

  for (int i = tid; i < NK_DIM / 4; i += 256) {
    float4 o;
    o.x = expf((buf[4 * i + 0] - m) * itau) * iz2;
    o.y = expf((buf[4 * i + 1] - m) * itau) * iz2;
    o.z = expf((buf[4 * i + 2] - m) * itau) * iz2;
    o.w = expf((buf[4 * i + 3] - m) * itau) * iz2;
    ((float4*)rp)[i] = o;
  }
}

// ---------- kernel 3: output = attn * values ----------
// attn tile (the 256MB stream): async double-buffered fp32 LDS.
// values tile (L2-resident, needs transpose): VGPR staging to bf16 LDS.
// Block tile 128(M) x 64(N), K-chunk 32. 8 waves: wave tile 32x32.

#define AV_STEPS (NK_DIM / 32)   // 128

__global__ __launch_bounds__(256) void edt_gemm_av(
    const float* __restrict__ A, const float* __restrict__ V,
    float* __restrict__ O) {
  __shared__ __align__(16) float  la[2][128][32];   // 32 KB
  __shared__ __align__(16) __bf16 lv[64][32];       //  4 KB, stored [n][k]

  const int tid  = threadIdx.x;
  const int lane = tid & 31;
  const int wave = tid >> 5;
  const int half = lane >> 4;
  const int l16  = lane & 15;

  const size_t bm0 = (size_t)blockIdx.y * 128;
  const size_t bn0 = (size_t)blockIdx.x * 64;
  const int wm0 = (wave >> 1) * 32;
  const int wn0 = (wave & 1) * 32;

  v8f acc[2][2] = {};

  auto stageA = [&](int b, int k0) {
    // 4 async b128 issues per thread => ASYNCcnt += 4 / wave
    for (int it = 0; it < 4; ++it) {
      const int idx = tid + it * 256;
      const int row = idx >> 3;
      const int col = (idx & 7) * 4;
      cp_async_f4(&la[b][row][col], A + (bm0 + row) * NK_DIM + k0 + col);
    }
  };

  stageA(0, 0);
  for (int kt = 0; kt < AV_STEPS; ++kt) {
    const int b = kt & 1;

    if (kt + 1 < AV_STEPS) stageA(b ^ 1, (kt + 1) * 32);

    // VGPR-stage values chunk kt (32k x 64n), transposed into [n][k] as bf16
    for (int it = 0; it < 2; ++it) {
      const int idx = tid + it * 256;          // 512 float4 slots
      const int kr  = idx >> 4;                // 16 float4 per 64-wide k-row
      const int col = (idx & 15) * 4;
      const float4 vv =
          *(const float4*)(V + (size_t)(kt * 32 + kr) * DV_DIM + bn0 + col);
      lv[col + 0][kr] = (__bf16)vv.x;
      lv[col + 1][kr] = (__bf16)vv.y;
      lv[col + 2][kr] = (__bf16)vv.z;
      lv[col + 3][kr] = (__bf16)vv.w;
    }

    if (kt + 1 < AV_STEPS) wait_async_le<4>(); else wait_async_le<0>();
    __syncthreads();

    v16bf af[2], bfr[2];
    for (int m = 0; m < 2; ++m) {
      const int r = wm0 + m * 16 + l16;
      af[m] = frag_from_f32(&la[b][r][half * 8], &la[b][r][16 + half * 8]);
    }
    for (int n = 0; n < 2; ++n) {
      const int c = wn0 + n * 16 + l16;
      bfr[n] = frag_from_bf16(&lv[c][half * 8], &lv[c][16 + half * 8]);
    }
    for (int m = 0; m < 2; ++m)
      for (int n = 0; n < 2; ++n)
        acc[m][n] = __builtin_amdgcn_wmma_f32_16x16x32_bf16(
            false, af[m], false, bfr[n], (short)0, acc[m][n], false, false);
    __syncthreads();
  }

  for (int m = 0; m < 2; ++m)
    for (int n = 0; n < 2; ++n) {
      const size_t r0 = bm0 + wm0 + m * 16 + half * 8;
      const size_t c  = bn0 + wn0 + n * 16 + l16;
      for (int r = 0; r < 8; ++r)
        O[(r0 + r) * DV_DIM + c] = acc[m][n][r];
    }
}

// ---------------- launcher ----------------

extern "C" void kernel_launch(void* const* d_in, const int* in_sizes, int n_in,
                              void* d_out, int out_size, void* d_ws, size_t ws_size,
                              hipStream_t stream) {
  (void)in_sizes; (void)n_in; (void)out_size; (void)d_ws; (void)ws_size;

  const float* q  = (const float*)d_in[0];
  const float* ks = (const float*)d_in[1];
  const float* vs = (const float*)d_in[2];
  const float* w1 = (const float*)d_in[3];
  const float* b1 = (const float*)d_in[4];
  const float* w2 = (const float*)d_in[5];
  const float* b2 = (const float*)d_in[6];

  // d_out layout: output (B*DV) | attn (B*NK) | entropy (B) | tau (B)
  float* out  = (float*)d_out;
  float* attn = out  + (size_t)B_DIM * DV_DIM;
  float* ent  = attn + (size_t)B_DIM * NK_DIM;
  float* tau  = ent  + B_DIM;

  edt_gemm_qk<<<dim3(NK_DIM / 128, B_DIM / 128), 256, 0, stream>>>(q, ks, attn);
  edt_row<<<B_DIM, 256, 0, stream>>>(attn, ent, tau, w1, b1, w2, b2);
  edt_gemm_av<<<dim3(DV_DIM / 64, B_DIM / 128), 256, 0, stream>>>(attn, vs, out);
}
